// Graph_SDE_1417339208186
// MI455X (gfx1250) — matrix operand order repaired and checked by reading, not compiled, not run on hardware
//
#include <hip/hip_runtime.h>
#include <hip/hip_bf16.h>

#define NN 10000
#define EE 160000

typedef __attribute__((ext_vector_type(16))) __bf16        v16bf;
typedef __attribute__((ext_vector_type(8)))  float         v8f;
typedef __attribute__((ext_vector_type(4)))  unsigned int  v4u;

union ABfrag { unsigned int u[8]; v4u q[2]; v16bf v; };

__device__ __forceinline__ unsigned short f2bf(float x) {
  unsigned int u = __float_as_uint(x);
  u += 0x7FFFu + ((u >> 16) & 1u);          // round-to-nearest-even
  return (unsigned short)(u >> 16);
}

__device__ __forceinline__ unsigned int pack2bf(float lo, float hi) {
  return (unsigned int)f2bf(lo) | ((unsigned int)f2bf(hi) << 16);
}

// IEEE float max via sign-split integer atomics (buffer init'd to -inf bits).
__device__ __forceinline__ void atomicMaxF(float* addr, float val) {
  if (val >= 0.0f) atomicMax((int*)addr, __float_as_int(val));
  else             atomicMin((unsigned int*)addr, __float_as_uint(val));
}

// ---------------------------------------------------------------------------
// Encoder: temb = silu(concat(sin,cos) @ teW + teb); h = x@encW+encb
// Writes Hcat bf16 [N,768] = [h | temb | emb_pcs]
// ---------------------------------------------------------------------------
__global__ void __launch_bounds__(256)
encode_kernel(const float* __restrict__ x, const float* __restrict__ t,
              const float* __restrict__ emb, const float* __restrict__ encW,
              const float* __restrict__ encb, const float* __restrict__ fw,
              const float* __restrict__ teW, const float* __restrict__ teb,
              unsigned short* __restrict__ Hcat) {
  __shared__ float sc[256];
  const int i = blockIdx.x;
  const int j = threadIdx.x;
  const float tv = t[i];
  if (j < 128) {
    float p = tv * fw[j] * 6.283185307179586f;
    sc[j] = sinf(p);
    sc[j + 128] = cosf(p);
  }
  __syncthreads();
  float acc = teb[j];
#pragma unroll 8
  for (int k = 0; k < 256; k++) acc += sc[k] * teW[k * 256 + j];
  float temb = acc * (1.0f / (1.0f + expf(-acc)));   // x*sigmoid(x)
  float he = encb[j];
#pragma unroll
  for (int d = 0; d < 3; d++) he += x[i * 3 + d] * encW[d * 256 + j];
  Hcat[i * 768 + j]       = f2bf(he);
  Hcat[i * 768 + 256 + j] = f2bf(temb);
  Hcat[i * 768 + 512 + j] = f2bf(emb[i * 256 + j]);
}

// ---------------------------------------------------------------------------
// Weight prep -> WMMA-B fragment-contiguous layout.
// For K-step s (32 K), half h (lane group), column n: the 8 dwords v=0..7 hold
// bf16 k-pairs (k, k+1) with k = s*32 + h*16 + v*2, stored consecutively, so a
// lane's whole B fragment is two 16B LDS loads.
//
// Node weights: W1 [1536,256] fp32 -> Wpk bf16 "[768,512]"
//   col j<256 : W1top - W1bot  (dst coefficient; bias added in GEMM epilogue)
//   col j>=256: W1bot          (src coefficient)
// dword index = s*8192 + h*4096 + n*8 + v      (24 steps x 8192 dwords)
// ---------------------------------------------------------------------------
__device__ __forceinline__ float w1comb(const float* __restrict__ W1, int k, int n) {
  return (n < 256) ? (W1[k * 256 + n] - W1[(768 + k) * 256 + n])
                   : W1[(768 + k) * 256 + (n - 256)];
}

__global__ void __launch_bounds__(256)
prep_w1_kernel(const float* __restrict__ W1, unsigned int* __restrict__ Wpk) {
  int idx = blockIdx.x * 256 + threadIdx.x;   // 196608 dwords total
  int s = idx >> 13;
  int r = idx & 8191;
  int h = r >> 12;
  int n = (r >> 3) & 511;
  int v = r & 7;
  int k = s * 32 + h * 16 + v * 2;
  Wpk[idx] = pack2bf(w1comb(W1, k, n), w1comb(W1, k + 1, n));
}

// Edge weights: W2 [256,256] fp32 -> Wpk2; dword idx = s*4096 + h*2048 + n*8 + v
__global__ void __launch_bounds__(256)
prep_w2_kernel(const float* __restrict__ W2, unsigned int* __restrict__ Wpk2) {
  int idx = blockIdx.x * 256 + threadIdx.x;   // 32768 dwords total
  int s = idx >> 12;
  int r = idx & 4095;
  int h = r >> 11;
  int n = (r >> 3) & 255;
  int v = r & 7;
  int k = s * 32 + h * 16 + v * 2;
  Wpk2[idx] = pack2bf(W2[k * 256 + n], W2[(k + 1) * 256 + n]);
}

__global__ void __launch_bounds__(256)
fill_kernel(unsigned int* __restrict__ p, int n, unsigned int v) {
  int i = blockIdx.x * 256 + threadIdx.x;
  if (i < n) p[i] = v;
}

// ---------------------------------------------------------------------------
// Node GEMM: PQ[N,512] fp32 = Hcat[N,768] bf16 @ Wpk (+b1 on cols<256)
// WG = 256 thr = 8 waves, tile 16(M) x 512(N); wave = 16x64 (4 accum tiles); K step 32.
// ---------------------------------------------------------------------------
__global__ void __launch_bounds__(256)
node_gemm_kernel(const unsigned short* __restrict__ Hcat,
                 const unsigned int* __restrict__ Wpk,
                 const float* __restrict__ b1, float* __restrict__ PQ) {
  __shared__ __align__(16) unsigned int ldsA[16 * 16];    // 16 rows x 32 bf16 (1 KB)
  __shared__ __align__(16) unsigned int ldsB[8192];       // one K-step panel (32 KB)
  const int t = threadIdx.x;
  const int w = t >> 5, lane = t & 31;
  const int half = lane >> 4, mrow = lane & 15;
  const int m_base = blockIdx.x * 16;
  const unsigned int* Hg = (const unsigned int*)Hcat;
  const v4u* Wg4 = (const v4u*)Wpk;
  v4u* ldsB4 = (v4u*)ldsB;
  const v4u* ldsA4 = (const v4u*)ldsA;
  const v4u* ldsB4c = (const v4u*)ldsB;
  v8f acc[4] = {};

  for (int kk = 0; kk < 768; kk += 32) {
    __syncthreads();
    { int row = t >> 4, cd = t & 15;
      ldsA[t] = Hg[(m_base + row) * 384 + (kk >> 1) + cd]; }
#pragma unroll
    for (int i = 0; i < 8; i++) ldsB4[i * 256 + t] = Wg4[kk * 64 + i * 256 + t];
    if (kk + 32 < 768)
      __builtin_prefetch((const void*)(Wg4 + (kk + 32) * 64 + t * 8), 0, 0);
    __syncthreads();

    ABfrag af;                                  // A frag: 2 x 16B LDS loads
    af.q[0] = ldsA4[mrow * 4 + half * 2];
    af.q[1] = ldsA4[mrow * 4 + half * 2 + 1];
#pragma unroll
    for (int nt = 0; nt < 4; nt++) {
      int n = w * 64 + nt * 16 + mrow;
      ABfrag bf;                                // B frag: 2 x 16B LDS loads
      bf.q[0] = ldsB4c[half * 1024 + n * 2];
      bf.q[1] = ldsB4c[half * 1024 + n * 2 + 1];
      acc[nt] = __builtin_amdgcn_wmma_f32_16x16x32_bf16(
          false, af.v, false, bf.v, (short)0, acc[nt], false, false);
    }
  }
#pragma unroll
  for (int nt = 0; nt < 4; nt++) {
    int col = w * 64 + nt * 16 + mrow;
    float bias = (col < 256) ? b1[col] : 0.0f;
#pragma unroll
    for (int j = 0; j < 8; j++) {
      int m = j + half * 8;                     // C/D VGPR-row layout
      PQ[(size_t)(m_base + m) * 512 + col] = acc[nt][j] + bias;
    }
  }
}

// ---------------------------------------------------------------------------
// Edge conv (layers 1,2): act = relu(P[dst]+Q[src]) [32 edges x 256] bf16 in LDS,
// WMMA with W2 [256,256], +b2, atomic segmented max into segmax[dst].
// WG = 8 waves: waves 0-3 -> edge tile 0, waves 4-7 -> tile 1; each wave 16x64.
// ---------------------------------------------------------------------------
__global__ void __launch_bounds__(256)
edge_conv_kernel(const float* __restrict__ PQ,
                 const unsigned int* __restrict__ Wpk2,
                 const float* __restrict__ b2, const int* __restrict__ ei,
                 float* __restrict__ segmax) {
  __shared__ __align__(16) unsigned short act16[32 * 256]; // 16 KB
  __shared__ __align__(16) unsigned int  ldsB[4096];       // 16 KB
  const int t = threadIdx.x;
  const int w = t >> 5, lane = t & 31;
  const int half = lane >> 4, mrow = lane & 15;
  const int tile = w >> 2, wn = (w & 3) * 64;
  const int e_base = blockIdx.x * 32;

  for (int i = 0; i < 32; i++) {                 // gather + relu + bf16
    int e = e_base + i;
    int s = ei[e], d = ei[EE + e];
    float v = PQ[(size_t)d * 512 + t] + PQ[(size_t)s * 512 + 256 + t];
    act16[i * 256 + t] = f2bf(fmaxf(v, 0.0f));
  }
  const v4u* actA4 = (const v4u*)act16;
  const v4u* Wg4 = (const v4u*)Wpk2;
  v4u* ldsB4 = (v4u*)ldsB;
  const v4u* ldsB4c = (const v4u*)ldsB;
  const int arow = (tile * 16 + mrow) * 32;
  v8f acc[4] = {};

  for (int kk = 0; kk < 256; kk += 32) {
    __syncthreads();
#pragma unroll
    for (int i = 0; i < 4; i++) ldsB4[i * 256 + t] = Wg4[kk * 32 + i * 256 + t];
    if (kk + 32 < 256)
      __builtin_prefetch((const void*)(Wg4 + (kk + 32) * 32 + t * 4), 0, 0);
    __syncthreads();

    ABfrag af;                                   // A frag: 2 x 16B LDS loads
    af.q[0] = actA4[arow + (kk >> 3) + half * 2];
    af.q[1] = actA4[arow + (kk >> 3) + half * 2 + 1];
#pragma unroll
    for (int nt = 0; nt < 4; nt++) {
      int n = wn + nt * 16 + mrow;
      ABfrag bf;                                 // B frag: 2 x 16B LDS loads
      bf.q[0] = ldsB4c[half * 512 + n * 2];
      bf.q[1] = ldsB4c[half * 512 + n * 2 + 1];
      acc[nt] = __builtin_amdgcn_wmma_f32_16x16x32_bf16(
          false, af.v, false, bf.v, (short)0, acc[nt], false, false);
    }
  }
  int dsts[8];
#pragma unroll
  for (int j = 0; j < 8; j++)
    dsts[j] = ei[EE + e_base + tile * 16 + (j + half * 8)];
#pragma unroll
  for (int nt = 0; nt < 4; nt++) {
    int col = wn + nt * 16 + mrow;
    float bias = b2[col];
#pragma unroll
    for (int j = 0; j < 8; j++)
      atomicMaxF(&segmax[(size_t)dsts[j] * 256 + col], acc[nt][j] + bias);
  }
}

// Post edge-conv (layers 1,2): zero non-finite, outer relu, write Hcat[:, :256] bf16.
__global__ void __launch_bounds__(256)
finalize12_kernel(const float* __restrict__ segmax, unsigned short* __restrict__ Hcat) {
  int i = blockIdx.x * 256 + threadIdx.x;    // N*256 exact
  float v = segmax[i];
  if (!(fabsf(v) < 3.0e38f)) v = 0.0f;
  v = fmaxf(v, 0.0f);
  int node = i >> 8, c = i & 255;
  Hcat[node * 768 + c] = f2bf(v);
}

// Layer 3 edge MLP: W2 is [256,3] -> one wave per edge, VALU + shuffle reduce.
__global__ void __launch_bounds__(256)
edge3_kernel(const float* __restrict__ PQ, const float* __restrict__ W2,
             const float* __restrict__ b2, const int* __restrict__ ei,
             float* __restrict__ segmax3) {
  const int w = threadIdx.x >> 5, lane = threadIdx.x & 31;
  const int e = blockIdx.x * 8 + w;
  const int s = ei[e], d = ei[EE + e];
  float a0 = 0.f, a1 = 0.f, a2 = 0.f;
#pragma unroll
  for (int c = lane; c < 256; c += 32) {
    float v = PQ[(size_t)d * 512 + c] + PQ[(size_t)s * 512 + 256 + c];
    v = fmaxf(v, 0.0f);
    a0 += v * W2[c * 3 + 0];
    a1 += v * W2[c * 3 + 1];
    a2 += v * W2[c * 3 + 2];
  }
#pragma unroll
  for (int off = 16; off > 0; off >>= 1) {
    a0 += __shfl_xor(a0, off);
    a1 += __shfl_xor(a1, off);
    a2 += __shfl_xor(a2, off);
  }
  if (lane == 0) {
    atomicMaxF(&segmax3[d * 3 + 0], a0 + b2[0]);
    atomicMaxF(&segmax3[d * 3 + 1], a1 + b2[1]);
    atomicMaxF(&segmax3[d * 3 + 2], a2 + b2[2]);
  }
}

__global__ void __launch_bounds__(256)
final_kernel(const float* __restrict__ segmax3, const float* __restrict__ t,
             float* __restrict__ out, int n3) {
  int i = blockIdx.x * 256 + threadIdx.x;
  if (i >= n3) return;
  float v = segmax3[i];
  if (!(fabsf(v) < 3.0e38f)) v = 0.0f;
  const float ln_s = 3.2188758248682006f;  // ln(25)
  float tv = t[i / 3];
  float stdv = sqrtf((expf(2.0f * tv * ln_s) - 1.0f) / (2.0f * ln_s));
  out[i] = v / (stdv + 1e-7f);
}

// ---------------------------------------------------------------------------
extern "C" void kernel_launch(void* const* d_in, const int* in_sizes, int n_in,
                              void* d_out, int out_size, void* d_ws, size_t ws_size,
                              hipStream_t stream) {
  const float* x    = (const float*)d_in[0];
  const int*   ei   = (const int*)d_in[1];
  const float* tt   = (const float*)d_in[2];
  const float* emb  = (const float*)d_in[3];
  const float* encW = (const float*)d_in[4];
  const float* encb = (const float*)d_in[5];
  const float* fw   = (const float*)d_in[6];
  const float* teW  = (const float*)d_in[7];
  const float* teb  = (const float*)d_in[8];
  const float* W1s[3] = {(const float*)d_in[9],  (const float*)d_in[13], (const float*)d_in[17]};
  const float* b1s[3] = {(const float*)d_in[10], (const float*)d_in[14], (const float*)d_in[18]};
  const float* W2s[3] = {(const float*)d_in[11], (const float*)d_in[15], (const float*)d_in[19]};
  const float* b2s[3] = {(const float*)d_in[12], (const float*)d_in[16], (const float*)d_in[20]};
  (void)in_sizes; (void)n_in; (void)out_size; (void)ws_size;

  char* ws = (char*)d_ws;
  size_t off = 0;
  auto take = [&](size_t bytes) {
    char* p = ws + off;
    off = (off + bytes + 255) & ~(size_t)255;
    return p;
  };
  unsigned short* Hcat = (unsigned short*)take((size_t)NN * 768 * 2);  // 15.4 MB
  float*          PQ   = (float*)take((size_t)NN * 512 * 4);           // 20.5 MB
  unsigned int*   Wpk  = (unsigned int*)take((size_t)768 * 512 * 2);   // 0.8 MB
  unsigned int*   Wpk2 = (unsigned int*)take((size_t)256 * 256 * 2);   // 0.13 MB
  float*          segm = (float*)take((size_t)NN * 256 * 4);           // 10.2 MB

  encode_kernel<<<NN, 256, 0, stream>>>(x, tt, emb, encW, encb, fw, teW, teb, Hcat);

  for (int l = 0; l < 2; l++) {
    prep_w1_kernel<<<768, 256, 0, stream>>>(W1s[l], Wpk);
    prep_w2_kernel<<<128, 256, 0, stream>>>(W2s[l], Wpk2);
    node_gemm_kernel<<<NN / 16, 256, 0, stream>>>(Hcat, Wpk, b1s[l], PQ);
    fill_kernel<<<(NN * 256 + 255) / 256, 256, 0, stream>>>((unsigned int*)segm, NN * 256, 0xFF800000u);
    edge_conv_kernel<<<EE / 32, 256, 0, stream>>>(PQ, Wpk2, b2s[l], ei, segm);
    finalize12_kernel<<<(NN * 256 + 255) / 256, 256, 0, stream>>>(segm, Hcat);
  }

  prep_w1_kernel<<<768, 256, 0, stream>>>(W1s[2], Wpk);
  node_gemm_kernel<<<NN / 16, 256, 0, stream>>>(Hcat, Wpk, b1s[2], PQ);
  fill_kernel<<<(NN * 3 + 255) / 256, 256, 0, stream>>>((unsigned int*)segm, NN * 3, 0xFF800000u);
  edge3_kernel<<<EE / 8, 256, 0, stream>>>(PQ, W2s[2], b2s[2], ei, segm);
  final_kernel<<<(NN * 3 + 255) / 256, 256, 0, stream>>>(segm, tt, (float*)d_out, NN * 3);
}
